// ConditionGeneration_52020643889648
// MI455X (gfx1250) — matrix operand order repaired and checked by reading, not compiled, not run on hardware
//
#include <hip/hip_runtime.h>

#define N_NUMF   100
#define N_CATG   100
#define N_COND   128
#define N_PERCAT 1000
#define OUT_N    200
#define EPS_GN   1e-5f
#define BT       32      // batch rows per cat block
#define GC       10      // group-staging chunk (100 % GC == 0)
#define WT_PITCH 136     // padded ushort pitch for W^T rows (272 B, 16B-aligned)
#define W_ELEMS  (N_COND * N_COND)          // 16384 per group
#define W_BYTES  (W_ELEMS * 2)              // 32 KB bf16 per group
#define W_CHUNKS (W_BYTES / 16)             // 2048 16-byte chunks per group

typedef __attribute__((ext_vector_type(16))) __bf16 v16bf;
typedef __attribute__((ext_vector_type(8)))  float  v8f;

struct U8x { unsigned int u[8]; };

__device__ __forceinline__ float fast_sigmoid(float x) {
  float e = __builtin_amdgcn_exp2f(-x * 1.44269504088896341f);
  return __builtin_amdgcn_rcpf(1.0f + e);
}

__device__ __forceinline__ unsigned short f2bf(float x) {
  unsigned int u = __builtin_bit_cast(unsigned int, x);
  u += 0x7FFFu + ((u >> 16) & 1u);         // round-to-nearest-even
  return (unsigned short)(u >> 16);
}

// ---------------- one-time: cat_W f32 [g][k][o] -> bf16 transposed [g][o][k]
__global__ void prep_w(const float* __restrict__ cat_W,
                       unsigned short* __restrict__ wbf) {
  __shared__ unsigned short sh[N_COND][N_COND + 1];
  const int g = blockIdx.x;
  const float* Wg = cat_W + (size_t)g * W_ELEMS;
  unsigned short* dst = wbf + (size_t)g * W_ELEMS;
  for (int i = threadIdx.x; i < W_ELEMS; i += blockDim.x) {
    int k = i >> 7, o = i & 127;
    sh[o][k] = f2bf(Wg[i]);                 // coalesced read
  }
  __syncthreads();
  for (int j = threadIdx.x; j < W_ELEMS; j += blockDim.x) {
    int o = j >> 7, k = j & 127;
    dst[j] = sh[o][k];                      // coalesced write, [o][k] layout
  }
}

// ---------------- num path: out[b, c, n] = sigmoid(x[b,n]*W[n,c]+b[n,c]), n<100
__global__ void num_kernel(const float* __restrict__ x_num,
                           const float* __restrict__ num_W,
                           const float* __restrict__ num_b,
                           float* __restrict__ out) {
  const int b = blockIdx.x;
  const float* xb = x_num + b * N_NUMF;
  float* ob = out + (size_t)b * (N_COND * OUT_N);
  for (int idx = threadIdx.x; idx < N_COND * N_NUMF; idx += blockDim.x) {
    int c = idx / N_NUMF;
    int n = idx - c * N_NUMF;
    float v = fast_sigmoid(xb[n] * num_W[n * N_COND + c] + num_b[n * N_COND + c]);
    ob[c * OUT_N + n] = v;
  }
}

// ---------------- cat path: gather -> groupnorm -> bf16 WMMA GEMM -> sigmoid
// 256 threads (8 waves), 32 batch rows, loops over all 100 groups; W tiles are
// async-DMA'd into a double-buffered LDS image, prefetched one group ahead.
__global__ void __launch_bounds__(256, 1)
cat_kernel(const int*   __restrict__ x_cat,
           const float* __restrict__ emb,
           const float* __restrict__ gn_gamma,
           const float* __restrict__ gn_beta,
           const float* __restrict__ cat_W,
           const float* __restrict__ cat_b,
           const unsigned short* __restrict__ wbf,
           const int use_async,
           float* __restrict__ out) {
  __shared__ unsigned short sh_h[BT][N_COND];              // bf16 h        (8 KB)
  __shared__ unsigned short sh_wt[2][N_COND][WT_PITCH];    // bf16 W^T x2  (68 KB)
  __shared__ float          sh_stage[BT][N_COND][GC];      // staged out  (160 KB)

  const int tid  = threadIdx.x;
  const int lane = tid & 31;
  const int wv   = tid >> 5;       // wave 0..7
  const int half = lane >> 4;      // 0/1
  const int l16  = lane & 15;
  const int b0   = blockIdx.x * BT;
  const int o_base = wv * 16;      // each wave owns one 16-wide N tile

  // prologue: kick off async DMA of W[0] into buffer 0
  if (use_async) {
    #pragma unroll
    for (int j = 0; j < 8; ++j) {
      int chunk = (wv * 8 + j) * 32 + lane;            // 0..2047, 16B each
      unsigned goff = (unsigned)chunk * 16u;
      unsigned ldsa = (unsigned)(unsigned long long)(&sh_wt[0][0][0]) +
                      (unsigned)((chunk >> 4) * (WT_PITCH * 2) + (chunk & 15) * 16);
      asm volatile("global_load_async_to_lds_b128 %0, %1, %2"
                   :: "v"(ldsa), "v"(goff), "s"(wbf) : "memory");
    }
  }

  for (int g = 0; g < N_CATG; ++g) {
    const int buf = g & 1;
    // -- gather + groupnorm: 4 rows per wave, 4 channels per lane --
    for (int rr = 0; rr < BT / 8; ++rr) {
      int r = wv * (BT / 8) + rr;
      int b = b0 + r;
      int eidx = x_cat[b * N_CATG + g] + g * N_PERCAT;
      float4 ev = ((const float4*)(emb + (size_t)eidx * N_COND))[lane];
      float s  = ev.x + ev.y + ev.z + ev.w;
      float s2 = ev.x * ev.x + ev.y * ev.y + ev.z * ev.z + ev.w * ev.w;
      #pragma unroll
      for (int off = 16; off > 0; off >>= 1) {
        s  += __shfl_xor(s, off, 32);
        s2 += __shfl_xor(s2, off, 32);
      }
      float mu   = s * (1.0f / N_COND);
      float var  = s2 * (1.0f / N_COND) - mu * mu;
      float rstd = __builtin_amdgcn_rsqf(var + EPS_GN);
      float4 gv = ((const float4*)(gn_gamma + g * N_COND))[lane];
      float4 bv = ((const float4*)(gn_beta  + g * N_COND))[lane];
      int c = lane * 4;
      sh_h[r][c + 0] = f2bf((ev.x - mu) * rstd * gv.x + bv.x);
      sh_h[r][c + 1] = f2bf((ev.y - mu) * rstd * gv.y + bv.y);
      sh_h[r][c + 2] = f2bf((ev.z - mu) * rstd * gv.z + bv.z);
      sh_h[r][c + 3] = f2bf((ev.w - mu) * rstd * gv.w + bv.w);
    }
    // -- W tile for this group --
    if (use_async) {
      if (g + 1 < N_CATG) {
        // prefetch next group into the other buffer, then wait for this group's
        // 8 older DMAs (in-order completion) before consuming it.
        const unsigned short* src = wbf + (size_t)(g + 1) * W_ELEMS;
        #pragma unroll
        for (int j = 0; j < 8; ++j) {
          int chunk = (wv * 8 + j) * 32 + lane;
          unsigned goff = (unsigned)chunk * 16u;
          unsigned ldsa = (unsigned)(unsigned long long)(&sh_wt[buf ^ 1][0][0]) +
                          (unsigned)((chunk >> 4) * (WT_PITCH * 2) + (chunk & 15) * 16);
          asm volatile("global_load_async_to_lds_b128 %0, %1, %2"
                       :: "v"(ldsa), "v"(goff), "s"(src) : "memory");
        }
        asm volatile("s_wait_asynccnt 0x8" ::: "memory");
      } else {
        asm volatile("s_wait_asynccnt 0x0" ::: "memory");
      }
    } else {
      // fallback: convert f32 W in-kernel (no workspace available)
      const float* Wg = cat_W + (size_t)g * W_ELEMS;
      for (int i = tid; i < W_ELEMS; i += 256) {
        int k = i >> 7, o = i & 127;
        sh_wt[buf][o][k] = f2bf(Wg[i]);
      }
    }
    __syncthreads();

    // -- GEMM: two 16-row M tiles per wave, K=128 in 4 steps of 32 --
    v8f acc0 = {}, acc1 = {};
    #pragma unroll
    for (int kt = 0; kt < 4; ++kt) {
      U8x bu;
      {
        const unsigned short* wr = &sh_wt[buf][o_base + l16][kt * 32 + half * 16];
        #pragma unroll
        for (int v = 0; v < 8; ++v)
          bu.u[v] = *(const unsigned int*)(wr + 2 * v);
      }
      v16bf bfr = __builtin_bit_cast(v16bf, bu);
      #pragma unroll
      for (int mt = 0; mt < 2; ++mt) {
        U8x au;
        const unsigned short* hr = sh_h[mt * 16 + l16];
        #pragma unroll
        for (int v = 0; v < 8; ++v) {
          int K = kt * 32 + half * 16 + v * 2;
          au.u[v] = *(const unsigned int*)(hr + K);
        }
        v16bf afr = __builtin_bit_cast(v16bf, au);
        if (mt == 0)
          acc0 = __builtin_amdgcn_wmma_f32_16x16x32_bf16(false, afr, false, bfr,
                                                         (short)0, acc0, false, false);
        else
          acc1 = __builtin_amdgcn_wmma_f32_16x16x32_bf16(false, afr, false, bfr,
                                                         (short)0, acc1, false, false);
      }
    }
    // -- epilogue: bias + sigmoid -> stage[m][o][g%GC] --
    {
      int o = o_base + l16;
      float bias = cat_b[g * N_COND + o];
      int gg = g % GC;
      #pragma unroll
      for (int r = 0; r < 8; ++r) {
        int m = r + 8 * half;
        sh_stage[m][o][gg]      = fast_sigmoid(acc0[r] + bias);
        sh_stage[16 + m][o][gg] = fast_sigmoid(acc1[r] + bias);
      }
    }
    __syncthreads();
    // -- flush staged groups as contiguous runs: out[b, o, 100+g] --
    if ((g % GC) == GC - 1) {
      int gbase = g - (GC - 1);
      for (int i = tid; i < BT * N_COND * GC; i += 256) {
        int gg = i % GC;
        int t  = i / GC;
        int o  = t & (N_COND - 1);
        int m  = t >> 7;
        out[(size_t)(b0 + m) * (N_COND * OUT_N) + o * OUT_N + N_NUMF + gbase + gg] =
            sh_stage[m][o][gg];
      }
      __syncthreads();
    }
  }
}

extern "C" void kernel_launch(void* const* d_in, const int* in_sizes, int n_in,
                              void* d_out, int out_size, void* d_ws, size_t ws_size,
                              hipStream_t stream) {
  const float* x_num    = (const float*)d_in[0];
  const int*   x_cat    = (const int*)  d_in[1];
  const float* emb      = (const float*)d_in[2];
  const float* num_W    = (const float*)d_in[3];
  const float* num_b    = (const float*)d_in[4];
  const float* gn_gamma = (const float*)d_in[5];
  const float* gn_beta  = (const float*)d_in[6];
  const float* cat_W    = (const float*)d_in[7];
  const float* cat_b    = (const float*)d_in[8];
  float* out = (float*)d_out;

  const int B = in_sizes[0] / N_NUMF;   // 8192
  const size_t need_ws = (size_t)N_CATG * W_ELEMS * sizeof(unsigned short);
  const int use_async = (ws_size >= need_ws) ? 1 : 0;
  unsigned short* wbf = (unsigned short*)d_ws;

  if (use_async)
    prep_w<<<dim3(N_CATG), dim3(256), 0, stream>>>(cat_W, wbf);
  num_kernel<<<dim3(B), dim3(256), 0, stream>>>(x_num, num_W, num_b, out);
  cat_kernel<<<dim3(B / BT), dim3(256), 0, stream>>>(x_cat, emb, gn_gamma, gn_beta,
                                                     cat_W, cat_b, wbf, use_async, out);
  (void)n_in; (void)out_size;
}
